// Qwen3MoeSparseMoeBlock_88235808129236
// MI455X (gfx1250) — compile-verified, matrix operand order checked by reading, not run on hardware
//
#include <hip/hip_runtime.h>

// ---------------------------------------------------------------------------
// Qwen3-style MoE block for MI455X (gfx1250): bf16 WMMA, fp32 accumulate.
// 128x64x32 macro-tiles, double-buffered LDS pipeline (1 barrier/iter),
// async global->LDS copies for bf16 A-tiles, v_cvt_pk_bf16_f32 for weights.
// ---------------------------------------------------------------------------

namespace {
constexpr int T_   = 2048;   // tokens
constexpr int H_   = 2048;   // hidden
constexpr int E_   = 64;     // experts
constexpr int I_   = 768;    // expert intermediate
constexpr int SI_  = 5632;   // shared-expert intermediate
constexpr int CAP_ = 512;    // slots per expert
constexpr int LDT  = 40;     // LDS tile row stride in bf16 elems (32 + 8 pad)
}

typedef __bf16 bf16x16 __attribute__((ext_vector_type(16)));
typedef float  floatx8 __attribute__((ext_vector_type(8)));

union Frag { bf16x16 v; unsigned u[8]; };

// Packed fp32x2 -> bf16x2 (single VALU op on CDNA5).
__device__ __forceinline__ unsigned cvt2bf(float lo, float hi) {
  unsigned r;
  asm("v_cvt_pk_bf16_f32 %0, %1, %2" : "=v"(r) : "v"(lo), "v"(hi));
  return r;
}
__device__ __forceinline__ unsigned short f2bf(float f) {
  return (unsigned short)cvt2bf(f, f);
}

// Async copy of 32 contiguous bytes global -> LDS (2 x b128, ASYNCcnt).
// INST_OFFSET applies to both the LDS and global addresses (ISA 15.18.3).
__device__ __forceinline__ void async_copy32(unsigned lds_off, const void* gptr) {
  asm volatile("global_load_async_to_lds_b128 %0, %1, off"
               :: "v"(lds_off), "v"(gptr) : "memory");
  asm volatile("global_load_async_to_lds_b128 %0, %1, off offset:16"
               :: "v"(lds_off), "v"(gptr) : "memory");
}
__device__ __forceinline__ void wait_async() {
  asm volatile("s_wait_asynccnt 0x0" ::: "memory");
}
__device__ __forceinline__ unsigned lds_offset(const void* p) {
  return (unsigned)(uintptr_t)p;   // generic LDS pointer: offset in low 32 bits
}

// Load a 16x32 bf16 fragment from an LDS tile (row-major, stride LDT).
// Per ISA 7.12.2: lanes 0-15 hold rows 0-15 / K 0-7 & 16-23 in VGPR0-7,
// lanes 16-31 hold rows 0-15 / K 8-15 & 24-31.
__device__ __forceinline__ void ld_frag(const unsigned short* __restrict__ p, Frag& f) {
  const int lane = threadIdx.x & 31;
  const int r    = lane & 15;
  const int kb   = (lane >> 4) << 3;
  const unsigned* q = (const unsigned*)(p + r * LDT + kb);
#pragma unroll
  for (int i = 0; i < 4; ++i) {
    f.u[i]     = q[i];       // K = kb + 2i, 2i+1
    f.u[4 + i] = q[8 + i];   // K = kb + 16 + 2i, 2i+1
  }
}

__device__ __forceinline__ floatx8 wmma_bf16(const Frag& a, const Frag& b, floatx8 c) {
  return __builtin_amdgcn_wmma_f32_16x16x32_bf16(
      false, a.v, false, b.v, (short)0, c, false, false);
}

// Stage a (2k x 4n) patch of fp32 B into transposed bf16 LDS tile.
__device__ __forceinline__ void stageB(unsigned short* lB, int n4, int k2,
                                       const float4& r0, const float4& r1) {
  unsigned* d = (unsigned*)&lB[n4 * LDT + 2 * k2];  // dword row stride 20
  d[0]  = cvt2bf(r0.x, r1.x);
  d[20] = cvt2bf(r0.y, r1.y);
  d[40] = cvt2bf(r0.z, r1.z);
  d[60] = cvt2bf(r0.w, r1.w);
}

// ---------------------------------------------------------------------------
// Kernel 1: convert hidden_states fp32 -> bf16 + zero expert counters
// ---------------------------------------------------------------------------
__global__ __launch_bounds__(256)
void prep_kernel(const float* __restrict__ x, unsigned* __restrict__ xb,
                 int* __restrict__ counts) {
  const int idx = blockIdx.x * 256 + threadIdx.x;          // 1 float4 each
  const float4 v = ((const float4*)x)[idx];
  xb[idx * 2 + 0] = cvt2bf(v.x, v.y);
  xb[idx * 2 + 1] = cvt2bf(v.z, v.w);
  if (blockIdx.x == 0 && threadIdx.x < E_) counts[threadIdx.x] = 0;
}

// ---------------------------------------------------------------------------
// Kernel 2: router (fp32 gate) + top-8 + slotting, plus sigmoid shared gate
// One block of 256 threads per token.
// ---------------------------------------------------------------------------
__global__ __launch_bounds__(256)
void router_kernel(const float* __restrict__ x, const float* __restrict__ gate_w,
                   const float* __restrict__ sg_w, int* __restrict__ counts,
                   int* __restrict__ slot_token, float* __restrict__ slot_weight,
                   float* __restrict__ sig_gate) {
  const int t   = blockIdx.x;
  const int tid = threadIdx.x;
  __shared__ float part[4 * E_];
  __shared__ float logits[E_];
  __shared__ float probs[E_];
  __shared__ float red[256];

  const float* xr = x + (size_t)t * H_;
  const int e = tid & 63;
  const int p = tid >> 6;                                   // 4 partials/expert
  const float4* gw4 = (const float4*)(gate_w + (size_t)e * H_ + p * 512);
  const float4* xp4 = (const float4*)(xr + p * 512);
  float s = 0.f;
#pragma unroll 4
  for (int h = 0; h < 128; ++h) {
    float4 a = xp4[h], b = gw4[h];
    s += a.x * b.x + a.y * b.y + a.z * b.z + a.w * b.w;
  }
  part[p * E_ + e] = s;

  float sgp = 0.f;
  for (int h = tid; h < H_; h += 256) sgp += xr[h] * sg_w[h];
  red[tid] = sgp;
  __syncthreads();

  if (tid < E_)
    logits[tid] = part[tid] + part[E_ + tid] + part[2 * E_ + tid] + part[3 * E_ + tid];
  for (int off = 128; off > 0; off >>= 1) {
    __syncthreads();
    if (tid < off) red[tid] += red[tid + off];
  }
  __syncthreads();

  if (tid == 0) {
    sig_gate[t] = 1.f / (1.f + __expf(-red[0]));
    float mx = logits[0];
    for (int i = 1; i < E_; ++i) mx = fmaxf(mx, logits[i]);
    for (int i = 0; i < E_; ++i) probs[i] = __expf(logits[i] - mx);
    // top-8; softmax denominator cancels under renormalization
    float wsum = 0.f;
    int   sel[8];
    float sw[8];
#pragma unroll
    for (int k = 0; k < 8; ++k) {
      int   bi = 0;
      float bv = probs[0];
      for (int i = 1; i < E_; ++i) {
        float v = probs[i];
        if (v > bv) { bv = v; bi = i; }
      }
      sel[k] = bi; sw[k] = bv; wsum += bv;
      probs[bi] = -1.f;
    }
    const float inv = 1.f / wsum;
#pragma unroll
    for (int k = 0; k < 8; ++k) {
      const int ex  = sel[k];
      const int pos = atomicAdd(&counts[ex], 1);
      if (pos < CAP_) {
        slot_token[ex * CAP_ + pos]  = t;
        slot_weight[ex * CAP_ + pos] = sw[k] * inv;
      }
    }
  }
}

// ---------------------------------------------------------------------------
// SwiGLU tile core:  act = silu(A@B1) * (A@B3), bf16 out.
// Block = 256 threads (8 waves); macro-tile 128x64, K-step 32.
// Double-buffered LDS, async A copies, one barrier per K-step.
// Each wave owns a 32x32 quadrant -> 8 WMMA per step.
// ---------------------------------------------------------------------------
__device__ __forceinline__ void swiglu_core(
    const unsigned short* __restrict__ Abase, const int* __restrict__ gather,
    int validRows, const float* __restrict__ B1, const float* __restrict__ B3,
    int bStride, unsigned short* __restrict__ Out, int oStride, int Ktot,
    int m0, int n0) {
  __shared__ __align__(16) unsigned short lA[2][128 * LDT];
  __shared__ __align__(16) unsigned short lB1[2][64 * LDT];  // stored as B^T (n,k)
  __shared__ __align__(16) unsigned short lB3[2][64 * LDT];
  const int tid = threadIdx.x;

  // A staging: 128 rows x 32 K; thread -> (row am, 16 consecutive K at ak)
  const int am = tid >> 1;
  const int ak = (tid & 1) << 4;
  int arow = m0 + am;
  if (gather) arow = (arow < validRows) ? gather[arow] : 0;
  const char* gA = (const char*)(Abase + (size_t)arow * H_ + ak);
  const unsigned lA0off = lds_offset(&lA[0][am * LDT + ak]);
  const unsigned lA1off = lds_offset(&lA[1][am * LDT + ak]);

  // B staging: thread -> (K rows 2k2 / 2k2+1, 4 consecutive N at n4)
  const int k2 = tid >> 4;            // 0..15
  const int n4 = (tid & 15) << 2;     // 0..60
  const float* bp1 = B1 + (size_t)(2 * k2) * bStride + n0 + n4;
  const float* bp3 = B3 + (size_t)(2 * k2) * bStride + n0 + n4;

  const int wave = tid >> 5;
  const int mw = (wave >> 1) << 5;    // 0,32,64,96
  const int nw = (wave & 1) << 1;     // 16-col tiles nw, nw+1

  floatx8 z8 = {0.f,0.f,0.f,0.f,0.f,0.f,0.f,0.f};
  floatx8 g00 = z8, g01 = z8, g10 = z8, g11 = z8;
  floatx8 u00 = z8, u01 = z8, u10 = z8, u11 = z8;

  const int steps = Ktot >> 5;

  // ---- prologue: stage step 0 into buffer 0 ----
  async_copy32(lA0off, gA);
  {
    float4 r0 = *(const float4*)bp1;
    float4 r1 = *(const float4*)(bp1 + bStride);
    float4 s0 = *(const float4*)bp3;
    float4 s1 = *(const float4*)(bp3 + bStride);
    stageB(lB1[0], n4, k2, r0, r1);
    stageB(lB3[0], n4, k2, s0, s1);
  }
  gA  += 64;
  bp1 += (size_t)32 * bStride;
  bp3 += (size_t)32 * bStride;
  wait_async();
  __syncthreads();

  int buf = 0;
  for (int s = 0; s < steps; ++s) {
    const bool has_next = (s + 1 < steps);
    float4 r0{}, r1{}, q0{}, q1{};
    if (has_next) {
      r0 = *(const float4*)bp1;
      r1 = *(const float4*)(bp1 + bStride);
      q0 = *(const float4*)bp3;
      q1 = *(const float4*)(bp3 + bStride);
      async_copy32(buf ? lA0off : lA1off, gA);
      gA  += 64;
      bp1 += (size_t)32 * bStride;
      bp3 += (size_t)32 * bStride;
      __builtin_prefetch(bp1, 0, 3);
      __builtin_prefetch(bp3, 0, 3);
    }

    Frag fa0, fa1, fb10, fb11, fb30, fb31;
    ld_frag(&lA[buf][mw * LDT], fa0);
    ld_frag(&lA[buf][(mw + 16) * LDT], fa1);
    ld_frag(&lB1[buf][(nw * 16) * LDT], fb10);
    ld_frag(&lB1[buf][((nw + 1) * 16) * LDT], fb11);
    ld_frag(&lB3[buf][(nw * 16) * LDT], fb30);
    ld_frag(&lB3[buf][((nw + 1) * 16) * LDT], fb31);
    g00 = wmma_bf16(fa0, fb10, g00);
    g01 = wmma_bf16(fa0, fb11, g01);
    g10 = wmma_bf16(fa1, fb10, g10);
    g11 = wmma_bf16(fa1, fb11, g11);
    u00 = wmma_bf16(fa0, fb30, u00);
    u01 = wmma_bf16(fa0, fb31, u01);
    u10 = wmma_bf16(fa1, fb30, u10);
    u11 = wmma_bf16(fa1, fb31, u11);

    if (has_next) {
      stageB(lB1[buf ^ 1], n4, k2, r0, r1);
      stageB(lB3[buf ^ 1], n4, k2, q0, q1);
      wait_async();
      __syncthreads();
    }
    buf ^= 1;
  }

  const int lane = tid & 31;
  const int cn = lane & 15;
  const int cm = (lane >> 4) << 3;
#pragma unroll
  for (int i = 0; i < 2; ++i) {
#pragma unroll
    for (int j = 0; j < 2; ++j) {
      floatx8 g = i ? (j ? g11 : g10) : (j ? g01 : g00);
      floatx8 u = i ? (j ? u11 : u10) : (j ? u01 : u00);
      const int ncol = n0 + (nw + j) * 16 + cn;
#pragma unroll
      for (int r = 0; r < 8; ++r) {
        const int m = m0 + mw + i * 16 + cm + r;
        const float gg  = g[r];
        const float act = (gg / (1.f + __expf(-gg))) * u[r];
        Out[(size_t)m * oStride + ncol] = f2bf(act);
      }
    }
  }
}

// ---------------------------------------------------------------------------
// Plain tile core: C = A(bf16) @ B(fp32->bf16), macro-tile 128x64.
// Same double-buffered pipeline; 4 WMMA per wave per step.
// ---------------------------------------------------------------------------
__device__ __forceinline__ void mm_core(
    const unsigned short* __restrict__ Abase, int aStride,
    const float* __restrict__ B, int bStride, int Ktot, int m0, int n0,
    floatx8& c00, floatx8& c01, floatx8& c10, floatx8& c11,
    int& mw_out, int& nw_out) {
  __shared__ __align__(16) unsigned short lA[2][128 * LDT];
  __shared__ __align__(16) unsigned short lB[2][64 * LDT];
  const int tid = threadIdx.x;

  const int am = tid >> 1;
  const int ak = (tid & 1) << 4;
  const char* gA = (const char*)(Abase + (size_t)(m0 + am) * aStride + ak);
  const unsigned lA0off = lds_offset(&lA[0][am * LDT + ak]);
  const unsigned lA1off = lds_offset(&lA[1][am * LDT + ak]);

  const int k2 = tid >> 4;
  const int n4 = (tid & 15) << 2;
  const float* bp = B + (size_t)(2 * k2) * bStride + n0 + n4;

  const int wave = tid >> 5;
  const int mw = (wave >> 1) << 5;
  const int nw = (wave & 1) << 1;

  floatx8 z8 = {0.f,0.f,0.f,0.f,0.f,0.f,0.f,0.f};
  c00 = z8; c01 = z8; c10 = z8; c11 = z8;

  const int steps = Ktot >> 5;

  async_copy32(lA0off, gA);
  {
    float4 r0 = *(const float4*)bp;
    float4 r1 = *(const float4*)(bp + bStride);
    stageB(lB[0], n4, k2, r0, r1);
  }
  gA += 64;
  bp += (size_t)32 * bStride;
  wait_async();
  __syncthreads();

  int buf = 0;
  for (int s = 0; s < steps; ++s) {
    const bool has_next = (s + 1 < steps);
    float4 r0{}, r1{};
    if (has_next) {
      r0 = *(const float4*)bp;
      r1 = *(const float4*)(bp + bStride);
      async_copy32(buf ? lA0off : lA1off, gA);
      gA += 64;
      bp += (size_t)32 * bStride;
      __builtin_prefetch(bp, 0, 3);
    }

    Frag fa0, fa1, fb0, fb1;
    ld_frag(&lA[buf][mw * LDT], fa0);
    ld_frag(&lA[buf][(mw + 16) * LDT], fa1);
    ld_frag(&lB[buf][(nw * 16) * LDT], fb0);
    ld_frag(&lB[buf][((nw + 1) * 16) * LDT], fb1);
    c00 = wmma_bf16(fa0, fb0, c00);
    c01 = wmma_bf16(fa0, fb1, c01);
    c10 = wmma_bf16(fa1, fb0, c10);
    c11 = wmma_bf16(fa1, fb1, c11);

    if (has_next) {
      stageB(lB[buf ^ 1], n4, k2, r0, r1);
      wait_async();
      __syncthreads();
    }
    buf ^= 1;
  }
  mw_out = mw;
  nw_out = nw;
}

// ---------------------------------------------------------------------------
// Kernel 3: shared-expert pass A:  act_s = silu(x@ws1) * (x@ws3)
// ---------------------------------------------------------------------------
__global__ __launch_bounds__(256)
void swiglu_shared_kernel(const unsigned short* __restrict__ xb,
                          const float* __restrict__ ws1,
                          const float* __restrict__ ws3,
                          unsigned short* __restrict__ act_s) {
  swiglu_core(xb, nullptr, T_, ws1, ws3, SI_, act_s, SI_, H_,
              blockIdx.y * 128, blockIdx.x * 64);
}

// ---------------------------------------------------------------------------
// Kernel 4: shared-expert pass B:  out = sigmoid_gate * (act_s @ ws2)
// ---------------------------------------------------------------------------
__global__ __launch_bounds__(256)
void out_shared_kernel(const unsigned short* __restrict__ act_s,
                       const float* __restrict__ ws2,
                       const float* __restrict__ sig_gate,
                       float* __restrict__ out) {
  const int n0 = blockIdx.x * 64, m0 = blockIdx.y * 128;
  floatx8 c00, c01, c10, c11;
  int mw, nw;
  mm_core(act_s, SI_, ws2, H_, SI_, m0, n0, c00, c01, c10, c11, mw, nw);
  const int lane = threadIdx.x & 31;
  const int cn = lane & 15;
  const int cm = (lane >> 4) << 3;
#pragma unroll
  for (int i = 0; i < 2; ++i) {
#pragma unroll
    for (int j = 0; j < 2; ++j) {
      floatx8 c = i ? (j ? c11 : c10) : (j ? c01 : c00);
      const int ncol = n0 + (nw + j) * 16 + cn;
#pragma unroll
      for (int r = 0; r < 8; ++r) {
        const int m = m0 + mw + i * 16 + cm + r;
        out[(size_t)m * H_ + ncol] = sig_gate[m] * c[r];
      }
    }
  }
}

// ---------------------------------------------------------------------------
// Kernel 5: expert pass A (gathered rows):  act_e = silu(Xe@w1) * (Xe@w3)
// ---------------------------------------------------------------------------
__global__ __launch_bounds__(256)
void swiglu_expert_kernel(const unsigned short* __restrict__ xb,
                          const float* __restrict__ w1,
                          const float* __restrict__ w3,
                          const int* __restrict__ slot_token,
                          const int* __restrict__ counts,
                          unsigned short* __restrict__ act_e) {
  const int e = blockIdx.z;
  int cnt = counts[e];
  cnt = cnt > CAP_ ? CAP_ : cnt;
  const int m0 = blockIdx.y * 128;
  if (m0 >= cnt) return;                       // whole block, uniform
  swiglu_core(xb, slot_token + e * CAP_, cnt,
              w1 + (size_t)e * H_ * I_, w3 + (size_t)e * H_ * I_, I_,
              act_e + (size_t)e * CAP_ * I_, I_, H_, m0, blockIdx.x * 64);
}

// ---------------------------------------------------------------------------
// Kernel 6: expert pass B:  out[token] += weight * (act_e @ w2)
// ---------------------------------------------------------------------------
__global__ __launch_bounds__(256)
void out_expert_kernel(const unsigned short* __restrict__ act_e,
                       const float* __restrict__ w2,
                       const int* __restrict__ slot_token,
                       const float* __restrict__ slot_weight,
                       const int* __restrict__ counts,
                       float* __restrict__ out) {
  const int e = blockIdx.z;
  int cnt = counts[e];
  cnt = cnt > CAP_ ? CAP_ : cnt;
  const int m0 = blockIdx.y * 128;
  if (m0 >= cnt) return;
  const int n0 = blockIdx.x * 64;
  floatx8 c00, c01, c10, c11;
  int mw, nw;
  mm_core(act_e + (size_t)e * CAP_ * I_, I_,
          w2 + (size_t)e * I_ * H_, H_, I_, m0, n0, c00, c01, c10, c11, mw, nw);
  const int lane = threadIdx.x & 31;
  const int cn = lane & 15;
  const int cm = (lane >> 4) << 3;
#pragma unroll
  for (int i = 0; i < 2; ++i) {
#pragma unroll
    for (int j = 0; j < 2; ++j) {
      floatx8 c = i ? (j ? c11 : c10) : (j ? c01 : c00);
      const int ncol = n0 + (nw + j) * 16 + cn;
#pragma unroll
      for (int r = 0; r < 8; ++r) {
        const int m = m0 + mw + i * 16 + cm + r;
        if (m < cnt) {
          const int   tok = slot_token[e * CAP_ + m];
          const float w   = slot_weight[e * CAP_ + m];
          atomicAdd(&out[(size_t)tok * H_ + ncol], w * c[r]);
        }
      }
    }
  }
}

// ---------------------------------------------------------------------------
// Host launcher
// ---------------------------------------------------------------------------
extern "C" void kernel_launch(void* const* d_in, const int* in_sizes, int n_in,
                              void* d_out, int out_size, void* d_ws, size_t ws_size,
                              hipStream_t stream) {
  const float* x      = (const float*)d_in[0];
  const float* gate_w = (const float*)d_in[1];
  const float* w1     = (const float*)d_in[2];
  const float* w3     = (const float*)d_in[3];
  const float* w2     = (const float*)d_in[4];
  const float* ws1    = (const float*)d_in[5];
  const float* ws3    = (const float*)d_in[6];
  const float* ws2    = (const float*)d_in[7];
  const float* sg_w   = (const float*)d_in[8];
  float* out = (float*)d_out;

  char* ws = (char*)d_ws;
  size_t o = 0;
  unsigned short* xb     = (unsigned short*)(ws + o); o += (size_t)T_ * H_ * 2;
  unsigned short* act_s  = (unsigned short*)(ws + o); o += (size_t)T_ * SI_ * 2;
  unsigned short* act_e  = (unsigned short*)(ws + o); o += (size_t)E_ * CAP_ * I_ * 2;
  int*   counts      = (int*)(ws + o);   o += (size_t)E_ * sizeof(int);
  int*   slot_token  = (int*)(ws + o);   o += (size_t)E_ * CAP_ * sizeof(int);
  float* slot_weight = (float*)(ws + o); o += (size_t)E_ * CAP_ * sizeof(float);
  float* sig_gate    = (float*)(ws + o); o += (size_t)T_ * sizeof(float);
  (void)ws_size; (void)in_sizes; (void)n_in; (void)out_size;

  // 1. fp32 -> bf16 conversion of hidden states + zero expert counters
  prep_kernel<<<(T_ * H_) / (256 * 4), 256, 0, stream>>>(x, (unsigned*)xb, counts);

  // 2. router + top-8 + slotting + sigmoid shared gate
  router_kernel<<<T_, 256, 0, stream>>>(x, gate_w, sg_w, counts, slot_token,
                                        slot_weight, sig_gate);

  // 3. shared-expert SwiGLU up-projection
  {
    dim3 g(SI_ / 64, T_ / 128);
    swiglu_shared_kernel<<<g, 256, 0, stream>>>(xb, ws1, ws3, act_s);
  }
  // 4. shared-expert down-projection (initializes d_out)
  {
    dim3 g(H_ / 64, T_ / 128);
    out_shared_kernel<<<g, 256, 0, stream>>>(act_s, ws2, sig_gate, out);
  }
  // 5. routed-expert SwiGLU up-projection (gathered rows)
  {
    dim3 g(I_ / 64, CAP_ / 128, E_);
    swiglu_expert_kernel<<<g, 256, 0, stream>>>(xb, w1, w3, slot_token, counts, act_e);
  }
  // 6. routed-expert down-projection + weighted scatter-add combine
  {
    dim3 g(H_ / 64, CAP_ / 128, E_);
    out_expert_kernel<<<g, 256, 0, stream>>>(act_e, w2, slot_token, slot_weight,
                                             counts, out);
  }
}
